// SpanAttention_36481452212227
// MI455X (gfx1250) — matrix operand order repaired and checked by reading, not compiled, not run on hardware
//
#include <hip/hip_runtime.h>
#include <stdint.h>

// ---------------------------------------------------------------------------
// Problem constants
// ---------------------------------------------------------------------------
#define B       8
#define S       1024
#define DIM     768
#define NHEAD   8
#define ADIM    96          // DIM / NHEAD
#define NEGV    (-1e10f)

typedef __attribute__((ext_vector_type(16))) __bf16 v16bf;
typedef __attribute__((ext_vector_type(8)))  __bf16 v8bf;
typedef __attribute__((ext_vector_type(8)))  float  v8f;

static __device__ __forceinline__ __bf16 f2bf(float f) {
    union { float f; uint32_t u; } v; v.f = f;
    uint32_t r = v.u + 0x7FFFu + ((v.u >> 16) & 1u);   // round-to-nearest-even
    uint16_t h = (uint16_t)(r >> 16);
    __bf16 out;
    __builtin_memcpy(&out, &h, 2);
    return out;
}

// ---------------------------------------------------------------------------
// Kernel 0: bulk f32 -> bf16 conversion (8 elements / thread, 16B stores)
// ---------------------------------------------------------------------------
__global__ void cvt_kernel(const float* __restrict__ src,
                           __bf16*      __restrict__ dst, int n8) {
    const int i = blockIdx.x * blockDim.x + threadIdx.x;
    if (i < n8) {
        const float4* sp = reinterpret_cast<const float4*>(src) + (size_t)2 * i;
        float4 a = sp[0], b = sp[1];
        v8bf o;
        o[0] = f2bf(a.x); o[1] = f2bf(a.y); o[2] = f2bf(a.z); o[3] = f2bf(a.w);
        o[4] = f2bf(b.x); o[5] = f2bf(b.y); o[6] = f2bf(b.z); o[7] = f2bf(b.w);
        reinterpret_cast<v8bf*>(dst)[i] = o;
    }
}

// ---------------------------------------------------------------------------
// Kernel 1: span attention weights  aw[b,s] = softmax_s( mask( X . w^T ) )
// One block per batch, 256 threads. (f32 inputs for fidelity.)
// ---------------------------------------------------------------------------
__global__ void aw_kernel(const float* __restrict__ X,
                          const int*   __restrict__ tags,
                          const float* __restrict__ ww,
                          float*       __restrict__ aw) {
    __shared__ float logit[S];
    __shared__ float red[256];
    const int b   = blockIdx.x;
    const int tid = threadIdx.x;

    const float4* wp = reinterpret_cast<const float4*>(ww);
    for (int r = tid; r < S; r += 256) {
        const float4* xp = reinterpret_cast<const float4*>(X + ((size_t)b * S + r) * DIM);
        float acc = 0.f;
        #pragma unroll 4
        for (int c = 0; c < DIM / 4; ++c) {
            float4 xv = xp[c], wv = wp[c];
            acc += xv.x * wv.x + xv.y * wv.y + xv.z * wv.z + xv.w * wv.w;
        }
        if (tags[b * S + r] == 0) acc = NEGV;
        logit[r] = acc;
    }
    __syncthreads();

    float m = -3.4e38f;
    for (int r = tid; r < S; r += 256) m = fmaxf(m, logit[r]);
    red[tid] = m; __syncthreads();
    for (int s = 128; s > 0; s >>= 1) {
        if (tid < s) red[tid] = fmaxf(red[tid], red[tid + s]);
        __syncthreads();
    }
    m = red[0]; __syncthreads();

    float ssum = 0.f;
    for (int r = tid; r < S; r += 256) {
        float e = __expf(logit[r] - m);
        logit[r] = e;
        ssum += e;
    }
    red[tid] = ssum; __syncthreads();
    for (int s = 128; s > 0; s >>= 1) {
        if (tid < s) red[tid] += red[tid + s];
        __syncthreads();
    }
    const float inv = 1.f / red[0];
    __syncthreads();
    for (int r = tid; r < S; r += 256) aw[b * S + r] = logit[r] * inv;
}

// ---------------------------------------------------------------------------
// Kernel 2: projections  P = X . W^T  (bf16 WMMA, f32 accumulate)
// All-bf16 operands, no in-loop conversion. Wave tile 16(M) x 64(N):
// one A fragment feeds 4 WMMAs per k-step. Block 256 thr = 8 waves (4M x 2N)
// -> block tile 64 x 128. Output head-split: plane[(b*8+h)][s][d], j = d*8+h.
// blockIdx.z selects {Q->bf16, K->bf16, V->f32}.
// ---------------------------------------------------------------------------
__global__ void proj_kernel(const __bf16* __restrict__ Xbf,
                            const __bf16* __restrict__ Wqbf,
                            const __bf16* __restrict__ Wkbf,
                            const __bf16* __restrict__ Wvbf,
                            __bf16* __restrict__ Qbf,
                            __bf16* __restrict__ Kbf,
                            float*  __restrict__ Vf) {
    const int mode = blockIdx.z;
    const __bf16* W = (mode == 0) ? Wqbf : (mode == 1) ? Wkbf : Wvbf;

    const int lane  = threadIdx.x & 31;
    const int wave  = threadIdx.x >> 5;
    const int laneM = lane & 15;
    const bool hi   = lane >= 16;

    const int m0 = blockIdx.x * 64  + (wave >> 1) * 16;
    const int n0 = blockIdx.y * 128 + (wave & 1) * 64;

    v8f acc[4] = {};

    for (int k0 = 0; k0 < DIM; k0 += 32) {
        // A fragment: row of X, K split per lane-half
        const int arow = m0 + laneM;
        const int kb   = k0 + (hi ? 16 : 0);
        v8bf lo = *reinterpret_cast<const v8bf*>(Xbf + (size_t)arow * DIM + kb);
        v8bf hh = *reinterpret_cast<const v8bf*>(Xbf + (size_t)arow * DIM + kb + 8);
        v16bf af;
        #pragma unroll
        for (int e = 0; e < 8; ++e) { af[e] = lo[e]; af[e + 8] = hh[e]; }

        const int koff = k0 + (hi ? 16 : 0);
        #pragma unroll
        for (int j = 0; j < 4; ++j) {
            const int nrow = n0 + j * 16 + laneM;          // B[k,n] = W[n,k]
            v16bf bf = *reinterpret_cast<const v16bf*>(W + (size_t)nrow * DIM + koff);
            acc[j] = __builtin_amdgcn_wmma_f32_16x16x32_bf16(false, af, false, bf, (short)0, acc[j], false, false);
        }
    }

    // store, head-split layout; mode is grid-uniform
    if (mode == 0) {
        #pragma unroll
        for (int j = 0; j < 4; ++j)
            #pragma unroll
            for (int r = 0; r < 8; ++r) {
                const int M = m0 + r + (hi ? 8 : 0);
                const int N = n0 + j * 16 + laneM;
                const size_t idx = (((size_t)(M >> 10) * NHEAD + (N & 7)) * S + (M & (S - 1))) * ADIM + (N >> 3);
                Qbf[idx] = f2bf(acc[j][r]);
            }
    } else if (mode == 1) {
        #pragma unroll
        for (int j = 0; j < 4; ++j)
            #pragma unroll
            for (int r = 0; r < 8; ++r) {
                const int M = m0 + r + (hi ? 8 : 0);
                const int N = n0 + j * 16 + laneM;
                const size_t idx = (((size_t)(M >> 10) * NHEAD + (N & 7)) * S + (M & (S - 1))) * ADIM + (N >> 3);
                Kbf[idx] = f2bf(acc[j][r]);
            }
    } else {
        #pragma unroll
        for (int j = 0; j < 4; ++j)
            #pragma unroll
            for (int r = 0; r < 8; ++r) {
                const int M = m0 + r + (hi ? 8 : 0);
                const int N = n0 + j * 16 + laneM;
                const size_t idx = (((size_t)(M >> 10) * NHEAD + (N & 7)) * S + (M & (S - 1))) * ADIM + (N >> 3);
                Vf[idx] = acc[j][r];
            }
    }
}

// ---------------------------------------------------------------------------
// Kernel 3: rel = (Qh Kh^T)/sqrt(96); softmax over heads per (q,k);
//           R[b,k,h] += aw[b,q] * p.   Grid (64 k-tiles, 8 batches), 8 waves.
// Heads-softmax is per-lane: all 8 head accumulators keep element (q,k) in
// the same lane/slot. Wave32 halves folded with shfl_xor(16).
// ---------------------------------------------------------------------------
__global__ void rel_accum_kernel(const __bf16* __restrict__ Qbf,
                                 const __bf16* __restrict__ Kbf,
                                 const float*  __restrict__ aw,
                                 float*        __restrict__ R) {
    const int b  = blockIdx.y;
    const int k0 = blockIdx.x * 16;

    const int lane  = threadIdx.x & 31;
    const int wave  = threadIdx.x >> 5;
    const int laneM = lane & 15;
    const bool hi   = lane >= 16;
    const float scale = 0.1020620726159658f;   // 1/sqrt(96)

    float racc[NHEAD];
    #pragma unroll
    for (int h = 0; h < NHEAD; ++h) racc[h] = 0.f;

    for (int t = 0; t < 8; ++t) {
        const int q0 = (t * 8 + wave) * 16;

        float awv[8];
        #pragma unroll
        for (int r = 0; r < 8; ++r)
            awv[r] = aw[b * S + q0 + r + (hi ? 8 : 0)];

        v8f acc[NHEAD];
        #pragma unroll
        for (int h = 0; h < NHEAD; ++h) {
            const __bf16* Qp = Qbf + (size_t)(b * NHEAD + h) * S * ADIM;
            const __bf16* Kp = Kbf + (size_t)(b * NHEAD + h) * S * ADIM;
            v8f a = {};
            #pragma unroll
            for (int kk = 0; kk < ADIM; kk += 32) {
                const int arow = q0 + laneM;
                const int kb   = kk + (hi ? 16 : 0);
                v8bf lo = *reinterpret_cast<const v8bf*>(Qp + (size_t)arow * ADIM + kb);
                v8bf hh = *reinterpret_cast<const v8bf*>(Qp + (size_t)arow * ADIM + kb + 8);
                v16bf af;
                #pragma unroll
                for (int e = 0; e < 8; ++e) { af[e] = lo[e]; af[e + 8] = hh[e]; }
                const int brow = k0 + laneM;
                const int dofs = kk + (hi ? 16 : 0);
                v16bf bf = *reinterpret_cast<const v16bf*>(Kp + (size_t)brow * ADIM + dofs);
                a = __builtin_amdgcn_wmma_f32_16x16x32_bf16(false, af, false, bf, (short)0, a, false, false);
            }
            acc[h] = a;
        }

        #pragma unroll
        for (int r = 0; r < 8; ++r) {
            float x[NHEAD];
            float m = -3.4e38f;
            #pragma unroll
            for (int h = 0; h < NHEAD; ++h) { x[h] = acc[h][r] * scale; m = fmaxf(m, x[h]); }
            float s = 0.f;
            #pragma unroll
            for (int h = 0; h < NHEAD; ++h) { x[h] = __expf(x[h] - m); s += x[h]; }
            const float wgt = awv[r] / s;
            #pragma unroll
            for (int h = 0; h < NHEAD; ++h) racc[h] += x[h] * wgt;
        }
    }

    #pragma unroll
    for (int h = 0; h < NHEAD; ++h) {
        float v = racc[h];
        v += __shfl_xor(v, 16, 32);
        if (lane < 16)
            atomicAdd(&R[((size_t)b * S + k0 + lane) * NHEAD + h], v);
    }
}

// ---------------------------------------------------------------------------
// Kernel 4: out[b, d*8+h] = sum_k R[b,k,h] * V[b,h,k,d]
// ---------------------------------------------------------------------------
__global__ void out_kernel(const float* __restrict__ R,
                           const float* __restrict__ Vf,
                           float*       __restrict__ out) {
    __shared__ float rsh[S];
    const int bh = blockIdx.x;
    const int b  = bh >> 3;
    const int h  = bh & 7;
    const int tid = threadIdx.x;

    for (int i = tid; i < S; i += blockDim.x)
        rsh[i] = R[((size_t)b * S + i) * NHEAD + h];
    __syncthreads();

    if (tid < ADIM) {
        const float* vp = Vf + (size_t)bh * S * ADIM + tid;
        float acc = 0.f;
        for (int k = 0; k < S; ++k)
            acc += rsh[k] * vp[(size_t)k * ADIM];
        out[b * DIM + tid * NHEAD + h] = acc;
    }
}

// ---------------------------------------------------------------------------
// Launch
// ---------------------------------------------------------------------------
extern "C" void kernel_launch(void* const* d_in, const int* in_sizes, int n_in,
                              void* d_out, int out_size, void* d_ws, size_t ws_size,
                              hipStream_t stream) {
    const float* X    = (const float*)d_in[0];   // [8,1024,768]
    const int*   tags = (const int*)  d_in[1];   // [8,1024]
    const float* ww   = (const float*)d_in[2];   // [1,768]
    const float* wq   = (const float*)d_in[3];   // [768,768]
    const float* wk   = (const float*)d_in[4];   // [768,768]
    const float* wv   = (const float*)d_in[5];   // [768,768]
    float* out = (float*)d_out;                  // [8,768]

    char* ws = (char*)d_ws;
    float*  aw   = (float*)ws;                      ws += (size_t)B * S * sizeof(float);
    float*  R    = (float*)ws;                      ws += (size_t)B * S * NHEAD * sizeof(float);
    __bf16* Qbf  = (__bf16*)ws;                     ws += (size_t)B * NHEAD * S * ADIM * sizeof(__bf16);
    __bf16* Kbf  = (__bf16*)ws;                     ws += (size_t)B * NHEAD * S * ADIM * sizeof(__bf16);
    float*  Vf   = (float*)ws;                      ws += (size_t)B * NHEAD * S * ADIM * sizeof(float);
    __bf16* Xbf  = (__bf16*)ws;                     ws += (size_t)B * S * DIM * sizeof(__bf16);
    __bf16* Wqbf = (__bf16*)ws;                     ws += (size_t)DIM * DIM * sizeof(__bf16);
    __bf16* Wkbf = (__bf16*)ws;                     ws += (size_t)DIM * DIM * sizeof(__bf16);
    __bf16* Wvbf = (__bf16*)ws;                     ws += (size_t)DIM * DIM * sizeof(__bf16);

    hipMemsetAsync(R, 0, (size_t)B * S * NHEAD * sizeof(float), stream);

    // f32 -> bf16 staging
    {
        const int nx = B * S * DIM / 8;       // 786432
        cvt_kernel<<<(nx + 255) / 256, 256, 0, stream>>>(X,  Xbf,  nx);
        const int nw = DIM * DIM / 8;         // 73728
        cvt_kernel<<<(nw + 255) / 256, 256, 0, stream>>>(wq, Wqbf, nw);
        cvt_kernel<<<(nw + 255) / 256, 256, 0, stream>>>(wk, Wkbf, nw);
        cvt_kernel<<<(nw + 255) / 256, 256, 0, stream>>>(wv, Wvbf, nw);
    }

    aw_kernel<<<B, 256, 0, stream>>>(X, tags, ww, aw);

    dim3 pgrid(S * B / 64, DIM / 128, 3);
    proj_kernel<<<pgrid, 256, 0, stream>>>(Xbf, Wqbf, Wkbf, Wvbf, Qbf, Kbf, Vf);

    dim3 rgrid(S / 16, B);
    rel_accum_kernel<<<rgrid, 256, 0, stream>>>(Qbf, Kbf, aw, R);

    out_kernel<<<B * NHEAD, 128, 0, stream>>>(R, Vf, out);
}
